// TransformerLayer_33200097199044
// MI455X (gfx1250) — matrix operand order repaired and checked by reading, not compiled, not run on hardware
//
#include <hip/hip_runtime.h>
#include <cstddef>
#include <cstdint>

typedef _Float16 v16h __attribute__((ext_vector_type(16)));
typedef _Float16 h8   __attribute__((ext_vector_type(8)));
typedef _Float16 h2   __attribute__((ext_vector_type(2)));
typedef float    v8f  __attribute__((ext_vector_type(8)));

#define WMMA_F16(a, b, c) \
  __builtin_amdgcn_wmma_f32_16x16x32_f16(false, (a), false, (b), (short)0, (c), false, false)

// ---------------------------------------------------------------------------
// f32 -> f16 conversion
// ---------------------------------------------------------------------------
__global__ void cvt_f32_f16(const float* __restrict__ in, _Float16* __restrict__ out, size_t n) {
  size_t i = (size_t)blockIdx.x * blockDim.x + threadIdx.x;
  if (i < n) out[i] = (_Float16)in[i];
}

// ---------------------------------------------------------------------------
// GEMM: C[M,N] = A[M,K] * W[N,K]^T + bias, f16 operands, f32 accumulate (WMMA)
// Block: 256 threads (8 waves), tile 128x128, K-step 32.
// Tiles staged global->LDS with CDNA5 async DMA (global_load_async_to_lds_b128,
// tracked by ASYNCcnt), so no VGPR round-trip for operand staging.
// Wave w: wm = w&3 (32-row subtile), wn = w>>2 (64-col subtile) -> 8 WMMA/K-step.
// ---------------------------------------------------------------------------
__global__ __launch_bounds__(256) void gemm_wmma(
    const _Float16* __restrict__ A, const _Float16* __restrict__ W,
    const float* __restrict__ bias, float* __restrict__ C32,
    _Float16* __restrict__ C16, int M, int N, int K, int act_gelu)
{
  __shared__ __align__(16) _Float16 As[128 * 32];
  __shared__ __align__(16) _Float16 Bs[128 * 32];
  const int tid  = threadIdx.x;
  const int lane = tid & 31, wave = tid >> 5;
  const int half = lane >> 4, ln = lane & 15;
  const int wm = wave & 3, wn = wave >> 2;
  const int m0 = blockIdx.x * 128, n0 = blockIdx.y * 128;

  // LDS byte offsets of the staging buffers (generic -> AS(3) -> int)
  const uint32_t ldsA = (uint32_t)(uintptr_t)(__attribute__((address_space(3))) _Float16*)As;
  const uint32_t ldsB = (uint32_t)(uintptr_t)(__attribute__((address_space(3))) _Float16*)Bs;

  v8f acc[2][4];
  {
    v8f z = {};
#pragma unroll
    for (int i = 0; i < 2; ++i)
#pragma unroll
      for (int j = 0; j < 4; ++j) acc[i][j] = z;
  }

  // async chunk map: chunk c in [0,512): row = c>>2, col8 = (c&3)*8 elems, LDS byte = c*16
  const int c0 = tid, c1 = tid + 256;
  const int r0 = c0 >> 2, r1 = c1 >> 2;
  const int e0 = (c0 & 3) * 8, e1 = (c1 & 3) * 8;
  const uint32_t lA0 = ldsA + (uint32_t)c0 * 16, lA1 = ldsA + (uint32_t)c1 * 16;
  const uint32_t lB0 = ldsB + (uint32_t)c0 * 16, lB1 = ldsB + (uint32_t)c1 * 16;

  for (int k0 = 0; k0 < K; k0 += 32) {
    const uint64_t gA0 = (uint64_t)(uintptr_t)(A + (size_t)(m0 + r0) * K + k0 + e0);
    const uint64_t gA1 = (uint64_t)(uintptr_t)(A + (size_t)(m0 + r1) * K + k0 + e1);
    const uint64_t gB0 = (uint64_t)(uintptr_t)(W + (size_t)(n0 + r0) * K + k0 + e0);
    const uint64_t gB1 = (uint64_t)(uintptr_t)(W + (size_t)(n0 + r1) * K + k0 + e1);
    asm volatile("global_load_async_to_lds_b128 %0, %1, off" :: "v"(lA0), "v"(gA0) : "memory");
    asm volatile("global_load_async_to_lds_b128 %0, %1, off" :: "v"(lA1), "v"(gA1) : "memory");
    asm volatile("global_load_async_to_lds_b128 %0, %1, off" :: "v"(lB0), "v"(gB0) : "memory");
    asm volatile("global_load_async_to_lds_b128 %0, %1, off" :: "v"(lB1), "v"(gB1) : "memory");
    asm volatile("s_wait_asynccnt 0x0" ::: "memory");
    __syncthreads();

    // A fragments (two 16-row subtiles), per ISA 16-bit A layout
    v16h a[2];
#pragma unroll
    for (int j = 0; j < 8; ++j) {
      const int ko = half * 8 + 2 * j + ((j >= 4) ? 8 : 0);
#pragma unroll
      for (int sm = 0; sm < 2; ++sm) {
        h2 p = *(const h2*)(As + (wm * 32 + sm * 16 + ln) * 32 + ko);
        a[sm][2 * j] = p[0]; a[sm][2 * j + 1] = p[1];
      }
    }
    // B fragments (four 16-col subtiles): Bmat[k][n] = W[n][k]
    v16h bf[4];
#pragma unroll
    for (int j = 0; j < 8; ++j) {
      const int ko = half * 8 + 2 * j + ((j >= 4) ? 8 : 0);
#pragma unroll
      for (int sn = 0; sn < 4; ++sn) {
        h2 p = *(const h2*)(Bs + (wn * 64 + sn * 16 + ln) * 32 + ko);
        bf[sn][2 * j] = p[0]; bf[sn][2 * j + 1] = p[1];
      }
    }
#pragma unroll
    for (int sm = 0; sm < 2; ++sm)
#pragma unroll
      for (int sn = 0; sn < 4; ++sn)
        acc[sm][sn] = WMMA_F16(a[sm], bf[sn], acc[sm][sn]);
    __syncthreads();
  }

  // epilogue: C row = r + 8*half within each 16-row subtile, col = lane&15
#pragma unroll
  for (int sm = 0; sm < 2; ++sm) {
#pragma unroll
    for (int r = 0; r < 8; ++r) {
      const int row = m0 + wm * 32 + sm * 16 + 8 * half + r;
      if (row < M) {
#pragma unroll
        for (int sn = 0; sn < 4; ++sn) {
          const int col = n0 + wn * 64 + sn * 16 + ln;
          float v = acc[sm][sn][r] + bias[col];
          if (act_gelu) v = 0.5f * v * (1.0f + erff(v * 0.7071067811865476f));
          if (C32) C32[(size_t)row * N + col] = v;
          if (C16) C16[(size_t)row * N + col] = (_Float16)v;
        }
      }
    }
  }
}

// ---------------------------------------------------------------------------
// Pack per-head f16 tensors [B,H,Lpad,64] from projected rows [B*L,768],
// applying 3D rotary PE (coords != nullptr) and zero-padding dims 48..63
// and rows L..Lpad-1 (so masked keys contribute exactly zero).
// ---------------------------------------------------------------------------
__global__ void pack_heads(const _Float16* __restrict__ X, const float* __restrict__ coords,
                           _Float16* __restrict__ Out, int L, int Lpad)
{
  const long long t = (long long)blockIdx.x * blockDim.x + threadIdx.x;
  const long long total = (long long)8 * Lpad * 16 * 32;
  if (t >= total) return;
  const int j = (int)(t & 31);
  long long r = t >> 5;
  const int h = (int)(r & 15); r >>= 4;
  const int l = (int)(r % Lpad);
  const int b = (int)(r / Lpad);
  _Float16* orow = Out + (((size_t)b * 16 + h) * (size_t)Lpad + l) * 64;

  if (l >= L) {  // padded key rows: force zero
    int d0 = (j < 24) ? ((j / 8) * 16 + 2 * (j % 8)) : (48 + (j - 24) * 2);
    orow[d0] = (_Float16)0.0f; orow[d0 + 1] = (_Float16)0.0f;
    return;
  }
  const _Float16* xr = X + ((size_t)b * L + l) * 768 + (size_t)h * 48;
  if (j < 24) {
    const int a = j / 8, i = j % 8, d0 = a * 16 + 2 * i;
    float x1 = (float)xr[d0], x2 = (float)xr[d0 + 1];
    if (coords) {
      const float c    = coords[((size_t)b * L + l) * 3 + a];
      const float invf = expf(-(float)(2 * i) * (9.210340371976184f / 16.0f)); // 10000^(-2i/16)
      const float ang = c * invf, cs = cosf(ang), sn = sinf(ang);
      orow[d0]     = (_Float16)(x1 * cs - x2 * sn);
      orow[d0 + 1] = (_Float16)(x1 * sn + x2 * cs);
    } else {
      orow[d0] = (_Float16)x1; orow[d0 + 1] = (_Float16)x2;
    }
  } else {
    const int d0 = 48 + (j - 24) * 2;
    orow[d0] = (_Float16)0.0f; orow[d0 + 1] = (_Float16)0.0f;
  }
}

// ---------------------------------------------------------------------------
// Flash attention: one wave per 16-row Q tile; head dim 48 padded to 64.
// Per 32-key block: 4 WMMA for scores, online softmax (width-16 shuffles
// matched to the C-matrix lane layout), P staged via per-wave LDS, 3 WMMA
// for P*V. Output written as f16 concat-head rows [B*Lq, 768].
// ---------------------------------------------------------------------------
__global__ __launch_bounds__(128) void flash_kernel(
    const _Float16* __restrict__ Q, const _Float16* __restrict__ Kp,
    const _Float16* __restrict__ Vp, _Float16* __restrict__ O,
    int Lq, int Lk, int Lkpad)
{
  __shared__ __align__(16) _Float16 Pl[4][16 * 32];
  const int wave = threadIdx.x >> 5, lane = threadIdx.x & 31;
  const int half = lane >> 4, ln = lane & 15;
  const int qt = blockIdx.x * 4 + wave;
  const int h = blockIdx.y, b = blockIdx.z;
  const int q0 = qt * 16;
  if (q0 >= Lq) return;  // no block-wide barriers used below

  const _Float16* qp = Q  + (((size_t)b * 16 + h) * (size_t)Lq + q0) * 64;
  const _Float16* kp = Kp + (((size_t)b * 16 + h) * (size_t)Lkpad) * 64;
  const _Float16* vp = Vp + (((size_t)b * 16 + h) * (size_t)Lkpad) * 64;

  // Q A-fragments for d = 0..31 and 32..63
  v16h qa0, qa1;
#pragma unroll
  for (int j = 0; j < 8; ++j) {
    const int ko = half * 8 + 2 * j + ((j >= 4) ? 8 : 0);
    h2 p0 = *(const h2*)(qp + ln * 64 + ko);
    h2 p1 = *(const h2*)(qp + ln * 64 + 32 + ko);
    qa0[2 * j] = p0[0]; qa0[2 * j + 1] = p0[1];
    qa1[2 * j] = p1[0]; qa1[2 * j + 1] = p1[1];
  }

  v8f acc0 = {}, acc1 = {}, acc2 = {};
  float mrow[8], lrow[8];
#pragma unroll
  for (int r = 0; r < 8; ++r) { mrow[r] = -1e30f; lrow[r] = 0.0f; }
  const float sc = 0.14433756729740643f;  // 1/sqrt(48)

  for (int kb = 0; kb < Lkpad; kb += 32) {
    // K B-fragments: Bmat[d][n] = K[key=n][d]; two key subtiles x two d-halves
    v16h k00, k01, k10, k11;
#pragma unroll
    for (int j = 0; j < 8; ++j) {
      const int dof = half * 8 + 2 * j + ((j >= 4) ? 8 : 0);
      h2 a0 = *(const h2*)(kp + ((size_t)(kb + ln)) * 64 + dof);
      h2 a1 = *(const h2*)(kp + ((size_t)(kb + ln)) * 64 + 32 + dof);
      h2 b0 = *(const h2*)(kp + ((size_t)(kb + 16 + ln)) * 64 + dof);
      h2 b1 = *(const h2*)(kp + ((size_t)(kb + 16 + ln)) * 64 + 32 + dof);
      k00[2 * j] = a0[0]; k00[2 * j + 1] = a0[1];
      k01[2 * j] = a1[0]; k01[2 * j + 1] = a1[1];
      k10[2 * j] = b0[0]; k10[2 * j + 1] = b0[1];
      k11[2 * j] = b1[0]; k11[2 * j + 1] = b1[1];
    }
    v8f c0 = {}, c1 = {};
    c0 = WMMA_F16(qa0, k00, c0);
    c0 = WMMA_F16(qa1, k01, c0);
    c1 = WMMA_F16(qa0, k10, c1);
    c1 = WMMA_F16(qa1, k11, c1);

    const bool msk0 = (kb + ln)      >= Lk;
    const bool msk1 = (kb + 16 + ln) >= Lk;
#pragma unroll
    for (int r = 0; r < 8; ++r) {
      float s0 = msk0 ? -1e30f : c0[r] * sc;
      float s1 = msk1 ? -1e30f : c1[r] * sc;
      float mx = fmaxf(s0, s1);
#pragma unroll
      for (int off = 8; off > 0; off >>= 1) mx = fmaxf(mx, __shfl_xor(mx, off, 16));
      const float mnew  = fmaxf(mrow[r], mx);
      const float alpha = expf(mrow[r] - mnew);
      const float p0 = expf(s0 - mnew);
      const float p1 = expf(s1 - mnew);
      float rs = p0 + p1;
#pragma unroll
      for (int off = 8; off > 0; off >>= 1) rs += __shfl_xor(rs, off, 16);
      lrow[r] = lrow[r] * alpha + rs;
      mrow[r] = mnew;
      acc0[r] *= alpha; acc1[r] *= alpha; acc2[r] *= alpha;
      // C-layout -> LDS (row-major 16x32) for A-fragment reload
      Pl[wave][(r + 8 * half) * 32 + ln]      = (_Float16)p0;
      Pl[wave][(r + 8 * half) * 32 + 16 + ln] = (_Float16)p1;
    }

    // P A-fragment (16x32 over keys-in-block)
    v16h pa;
#pragma unroll
    for (int j = 0; j < 8; ++j) {
      const int ko = half * 8 + 2 * j + ((j >= 4) ? 8 : 0);
      h2 p = *(const h2*)(&Pl[wave][ln * 32 + ko]);
      pa[2 * j] = p[0]; pa[2 * j + 1] = p[1];
    }
    // V B-fragments (keys x dim), three 16-col subtiles cover dims 0..47
    v16h vb0, vb1, vb2;
#pragma unroll
    for (int j = 0; j < 8; ++j) {
      const size_t k = (size_t)(kb + half * 8 + 2 * j + ((j >= 4) ? 8 : 0));
      const _Float16* vr0 = vp + k * 64;
      const _Float16* vr1 = vp + (k + 1) * 64;
      vb0[2 * j] = vr0[ln];      vb0[2 * j + 1] = vr1[ln];
      vb1[2 * j] = vr0[16 + ln]; vb1[2 * j + 1] = vr1[16 + ln];
      vb2[2 * j] = vr0[32 + ln]; vb2[2 * j + 1] = vr1[32 + ln];
    }
    acc0 = WMMA_F16(pa, vb0, acc0);
    acc1 = WMMA_F16(pa, vb1, acc1);
    acc2 = WMMA_F16(pa, vb2, acc2);
  }

  const size_t orow0 = (size_t)b * Lq + q0;
#pragma unroll
  for (int r = 0; r < 8; ++r) {
    const float inv = 1.0f / lrow[r];
    const size_t ro = (orow0 + r + 8 * half) * 768 + (size_t)h * 48;
    O[ro + ln]      = (_Float16)(acc0[r] * inv);
    O[ro + 16 + ln] = (_Float16)(acc1[r] * inv);
    O[ro + 32 + ln] = (_Float16)(acc2[r] * inv);
  }
}

// ---------------------------------------------------------------------------
// out = LayerNorm(X + Y) * g + be ; one block (256 thr) per 768-wide row.
// ---------------------------------------------------------------------------
__global__ __launch_bounds__(256) void add_ln_kernel(
    const float* __restrict__ X, const float* __restrict__ Y,
    const float* __restrict__ g, const float* __restrict__ be,
    float* __restrict__ O32, _Float16* __restrict__ O16)
{
  __shared__ float red[256];
  const int row = blockIdx.x, t = threadIdx.x;
  const size_t base = (size_t)row * 768;
  float v0 = X[base + t]       + Y[base + t];
  float v1 = X[base + t + 256] + Y[base + t + 256];
  float v2 = X[base + t + 512] + Y[base + t + 512];
  red[t] = v0 + v1 + v2; __syncthreads();
#pragma unroll
  for (int off = 128; off > 0; off >>= 1) { if (t < off) red[t] += red[t + off]; __syncthreads(); }
  const float mean = red[0] * (1.0f / 768.0f); __syncthreads();
  const float d0 = v0 - mean, d1 = v1 - mean, d2 = v2 - mean;
  red[t] = d0 * d0 + d1 * d1 + d2 * d2; __syncthreads();
#pragma unroll
  for (int off = 128; off > 0; off >>= 1) { if (t < off) red[t] += red[t + off]; __syncthreads(); }
  const float rstd = rsqrtf(red[0] * (1.0f / 768.0f) + 1e-5f);
  const float o0 = d0 * rstd * g[t]       + be[t];
  const float o1 = d1 * rstd * g[t + 256] + be[t + 256];
  const float o2 = d2 * rstd * g[t + 512] + be[t + 512];
  O32[base + t] = o0; O32[base + t + 256] = o1; O32[base + t + 512] = o2;
  if (O16) {
    O16[base + t] = (_Float16)o0; O16[base + t + 256] = (_Float16)o1; O16[base + t + 512] = (_Float16)o2;
  }
}

// ---------------------------------------------------------------------------
// Host orchestration
// ---------------------------------------------------------------------------
extern "C" void kernel_launch(void* const* d_in, const int* in_sizes, int n_in,
                              void* d_out, int out_size, void* d_ws, size_t ws_size,
                              hipStream_t stream)
{
  (void)in_sizes; (void)n_in; (void)out_size; (void)ws_size;
  const int NT = 8192;  // B*S tokens

  const float* f_src  = (const float*)d_in[0];
  const float* f_csrc = (const float*)d_in[1];
  const float* f_text = (const float*)d_in[2];
  const float* f_perc = (const float*)d_in[3];
  const float* f_ccam = (const float*)d_in[4];

  char* wp = (char*)d_ws;
  auto carve = [&](size_t bytes) -> char* {
    char* r = wp; wp += (bytes + 255) & ~(size_t)255; return r;
  };

  // f16 weight copies: self/text/perc {wq,wk,wv,wo} then w1, w2
  static const int    widx[14] = {5, 7, 9, 11, 13, 15, 17, 19, 21, 23, 25, 27, 29, 31};
  static const size_t wcnt[14] = {589824, 589824, 589824, 589824, 589824, 589824, 589824,
                                  589824, 589824, 589824, 589824, 589824, 2359296, 2359296};
  _Float16* w16[14];
  for (int i = 0; i < 14; ++i) w16[i] = (_Float16*)carve(wcnt[i] * sizeof(_Float16));

  _Float16* src16  = (_Float16*)carve((size_t)NT * 768 * 2);
  _Float16* text16 = (_Float16*)carve((size_t)616 * 768 * 2);
  _Float16* perc16 = (_Float16*)carve((size_t)2048 * 768 * 2);
  _Float16* qbuf   = (_Float16*)carve((size_t)NT * 768 * 2);
  _Float16* kbuf   = (_Float16*)carve((size_t)NT * 768 * 2);
  _Float16* vbuf   = (_Float16*)carve((size_t)NT * 768 * 2);
  _Float16* qpad   = (_Float16*)carve((size_t)8 * 16 * 1024 * 64 * 2);
  _Float16* kpad   = (_Float16*)carve((size_t)8 * 16 * 1024 * 64 * 2);
  _Float16* vpad   = (_Float16*)carve((size_t)8 * 16 * 1024 * 64 * 2);
  _Float16* attn16 = (_Float16*)carve((size_t)NT * 768 * 2);
  float*    proj32 = (float*)carve((size_t)NT * 768 * 4);
  float*    strm32 = (float*)carve((size_t)NT * 768 * 4);
  _Float16* strm16 = (_Float16*)carve((size_t)NT * 768 * 2);
  _Float16* h16    = (_Float16*)carve((size_t)NT * 3072 * 2);

  auto cvt = [&](const void* in, _Float16* out, size_t n) {
    cvt_f32_f16<<<dim3((unsigned)((n + 255) / 256)), 256, 0, stream>>>((const float*)in, out, n);
  };
  for (int i = 0; i < 14; ++i) cvt(d_in[widx[i]], w16[i], wcnt[i]);
  cvt(f_src,  src16,  (size_t)NT * 768);
  cvt(f_text, text16, (size_t)616 * 768);
  cvt(f_perc, perc16, (size_t)2048 * 768);

  auto gemm = [&](const _Float16* A, const _Float16* W, const float* bias,
                  float* C32, _Float16* C16, int M, int N, int K, int gelu) {
    gemm_wmma<<<dim3((M + 127) / 128, N / 128), 256, 0, stream>>>(A, W, bias, C32, C16, M, N, K, gelu);
  };
  auto pack = [&](const _Float16* X, const float* coords, _Float16* Out, int L, int Lpad) {
    size_t tot = (size_t)8 * Lpad * 16 * 32;
    pack_heads<<<dim3((unsigned)((tot + 255) / 256)), 256, 0, stream>>>(X, coords, Out, L, Lpad);
  };
  auto lnorm = [&](const float* X, const float* Y, int gi, int bi, float* O32, _Float16* O16) {
    add_ln_kernel<<<NT, 256, 0, stream>>>(X, Y, (const float*)d_in[gi], (const float*)d_in[bi], O32, O16);
  };
  auto attn_stage = [&](const _Float16* xq, const _Float16* xkv, int Mkv,
                        int Lk, int Lkpad, const float* ck, int pw, int pb) {
    gemm(xq,  w16[pw + 0], (const float*)d_in[pb + 1], nullptr, qbuf, NT,  768, 768, 0);
    gemm(xkv, w16[pw + 1], (const float*)d_in[pb + 3], nullptr, kbuf, Mkv, 768, 768, 0);
    gemm(xkv, w16[pw + 2], (const float*)d_in[pb + 5], nullptr, vbuf, Mkv, 768, 768, 0);
    pack(qbuf, f_csrc,  qpad, 1024, 1024);
    pack(kbuf, ck,      kpad, Lk, Lkpad);
    pack(vbuf, nullptr, vpad, Lk, Lkpad);
    flash_kernel<<<dim3(16, 16, 8), 128, 0, stream>>>(qpad, kpad, vpad, attn16, 1024, Lk, Lkpad);
    gemm(attn16, w16[pw + 3], (const float*)d_in[pb + 7], proj32, nullptr, NT, 768, 768, 0);
  };

  // Stage 1: self-attention (rotary on Q and K)
  attn_stage(src16, src16, NT, 1024, 1024, f_csrc, 0, 5);
  lnorm(f_src, proj32, 33, 34, strm32, strm16);          // g1, be1
  // Stage 2: cross-attention to text (K coords are zeros -> identity rotation)
  attn_stage(strm16, text16, 616, 77, 96, nullptr, 4, 13);
  lnorm(strm32, proj32, 37, 38, strm32, strm16);         // g3, be3
  // Stage 3: cross-attention to perceiver (rotary on K with camera coords)
  attn_stage(strm16, perc16, 2048, 256, 256, f_ccam, 8, 21);
  lnorm(strm32, proj32, 39, 40, strm32, strm16);         // g4, be4
  // FFN: GELU fused into GEMM1 epilogue (f16 hidden), GEMM2 back to f32
  gemm(strm16, w16[12], (const float*)d_in[30], nullptr, h16, NT, 3072, 768, 1);
  gemm(h16,    w16[13], (const float*)d_in[32], proj32, nullptr, NT, 768, 3072, 0);
  lnorm(strm32, proj32, 35, 36, (float*)d_out, nullptr); // g2, be2 -> final output
}
